// HFCAM_25357486916394
// MI455X (gfx1250) — matrix operand order repaired and checked by reading, not compiled, not run on hardware
//
#include <hip/hip_runtime.h>

// ---- problem constants (B,H,W,C from the reference) ----
#define BB 8
#define HH 128
#define WW 128
#define CC 256
#define NN (HH * WW)   // 16384
#define KSPLIT 4
#define KSLICE (NN / KSPLIT)   // 4096

typedef __attribute__((ext_vector_type(16))) __bf16 v16bf;
typedef __attribute__((ext_vector_type(8)))  __bf16 v8bf;
typedef __attribute__((ext_vector_type(8)))  float  v8f;

// load a 16-bit A-fragment (per-lane 2x16B contiguous halves)
__device__ __forceinline__ v16bf load_afrag(const __bf16* p) {
    v8bf lo = *(const v8bf*)(p);
    v8bf hi = *(const v8bf*)(p + 16);
    return __builtin_shufflevector(lo, hi,
            0,1,2,3,4,5,6,7,8,9,10,11,12,13,14,15);
}

// ---------------------------------------------------------------------------
// Kernel 1: x (B,N,C) f32  ->  XT (B,C,N) bf16   (LDS-tiled transpose+convert)
// grid: (N/64, C/64, B), block: 256
// ---------------------------------------------------------------------------
__global__ void ca_transpose_cvt(const float* __restrict__ x,
                                 __bf16* __restrict__ XT) {
    __shared__ __bf16 lt[64][64 + 8];   // [c][n], padded
    const int nb = blockIdx.x * 64;
    const int cb = blockIdx.y * 64;
    const int b  = blockIdx.z;
    const int t  = threadIdx.x;
    const int i  = t >> 4;           // 0..15 : n offset
    const int j4 = (t & 15) * 4;     // 0..60 : c offset (float4)

    const float* xb = x + (size_t)b * NN * CC;
    #pragma unroll
    for (int rr = 0; rr < 4; ++rr) {
        const int n = i + rr * 16;
        float4 f = *(const float4*)(xb + (size_t)(nb + n) * CC + cb + j4);
        lt[j4 + 0][n] = (__bf16)f.x;
        lt[j4 + 1][n] = (__bf16)f.y;
        lt[j4 + 2][n] = (__bf16)f.z;
        lt[j4 + 3][n] = (__bf16)f.w;
    }
    __syncthreads();

    __bf16* XTb = XT + (size_t)b * CC * NN;
    #pragma unroll
    for (int rr = 0; rr < 4; ++rr) {
        const int c = i + rr * 16;
        uint2 d = *(const uint2*)&lt[c][j4];            // 4 bf16 = 8 bytes
        *(uint2*)(XTb + (size_t)(cb + c) * NN + nb + j4) = d;
    }
}

// ---------------------------------------------------------------------------
// Kernel 2: split-K partial Gram: Spart[ks][b] += X_b^T X_b over K-slice
// one wave computes a 32x64 output block (2x4 tiles, 8 accumulators).
// waves = B * 8 ti-pairs * 4 tj-quads * KSPLIT = 1024 -> 128 blocks x 256 thr
// ---------------------------------------------------------------------------
__global__ void ca_gram_wmma(const __bf16* __restrict__ XT,
                             float* __restrict__ Spart) {
    const int w    = (blockIdx.x * blockDim.x + threadIdx.x) >> 5; // 0..1023
    const int lane = threadIdx.x & 31;
    const int ks   = w & 3;
    const int tjq  = (w >> 2) & 3;   // 0..3  -> 64 columns
    const int tip  = (w >> 4) & 7;   // 0..7  -> 32 rows
    const int b    = w >> 7;

    const __bf16* Xb = XT + (size_t)b * CC * NN;
    const int lo16 = lane & 15;
    const int hi   = lane >> 4;

    // A fragments: rows tip*32 + {0,16} + lo16, per-lane 16 halves at hi*16
    const __bf16* pA0 = Xb + (size_t)(tip * 32 + lo16) * NN + hi * 16;
    const __bf16* pA1 = pA0 + (size_t)16 * NN;
    // B fragments: cols tjq*64 + tj*16 + lo16, per-lane 16 halves at hi*16
    const __bf16* pB0 = Xb + (size_t)(tjq * 64 + lo16) * NN + hi * 16;

    v8f acc[2][4] = {};
    const int kbeg = ks * KSLICE;
    const int kend = kbeg + KSLICE;
    for (int k0 = kbeg; k0 < kend; k0 += 32) {
        v16bf a0 = *(const v16bf*)(pA0 + k0);
        v16bf a1 = *(const v16bf*)(pA1 + k0);
        v16bf b0 = *(const v16bf*)(pB0 + k0);
        v16bf b1 = *(const v16bf*)(pB0 + 16 * NN + k0);
        v16bf b2 = *(const v16bf*)(pB0 + 32 * NN + k0);
        v16bf b3 = *(const v16bf*)(pB0 + 48 * NN + k0);
        acc[0][0] = __builtin_amdgcn_wmma_f32_16x16x32_bf16(false, a0, false, b0, (short)0, acc[0][0], false, false);
        acc[0][1] = __builtin_amdgcn_wmma_f32_16x16x32_bf16(false, a0, false, b1, (short)0, acc[0][1], false, false);
        acc[0][2] = __builtin_amdgcn_wmma_f32_16x16x32_bf16(false, a0, false, b2, (short)0, acc[0][2], false, false);
        acc[0][3] = __builtin_amdgcn_wmma_f32_16x16x32_bf16(false, a0, false, b3, (short)0, acc[0][3], false, false);
        acc[1][0] = __builtin_amdgcn_wmma_f32_16x16x32_bf16(false, a1, false, b0, (short)0, acc[1][0], false, false);
        acc[1][1] = __builtin_amdgcn_wmma_f32_16x16x32_bf16(false, a1, false, b1, (short)0, acc[1][1], false, false);
        acc[1][2] = __builtin_amdgcn_wmma_f32_16x16x32_bf16(false, a1, false, b2, (short)0, acc[1][2], false, false);
        acc[1][3] = __builtin_amdgcn_wmma_f32_16x16x32_bf16(false, a1, false, b3, (short)0, acc[1][3], false, false);
    }

    // C/D layout: VGPR r -> M = r + 8*hi, N = lo16
    float* Sp = Spart + (size_t)ks * (BB * CC * CC) + ((size_t)b << 16);
    #pragma unroll
    for (int ti = 0; ti < 2; ++ti) {
        #pragma unroll
        for (int tj = 0; tj < 4; ++tj) {
            #pragma unroll
            for (int r = 0; r < 8; ++r) {
                const int m = tip * 32 + ti * 16 + r + hi * 8;
                const int n = tjq * 64 + tj * 16 + lo16;
                Sp[(size_t)m * CC + n] = acc[ti][tj][r];
            }
        }
    }
}

// ---------------------------------------------------------------------------
// Kernel 3: sum split-K partials + row softmax -> MT (b, d, c) in bf16
// grid: B*256 rows, block: 256
// ---------------------------------------------------------------------------
__global__ void ca_softmax_rows(const float* __restrict__ Spart,
                                __bf16* __restrict__ MT) {
    const int rowg = blockIdx.x;          // b*256 + c
    const int b = rowg >> 8;
    const int c = rowg & 255;
    const int t = threadIdx.x;

    const size_t stride = (size_t)BB * CC * CC;
    float v = 0.f;
    #pragma unroll
    for (int ks = 0; ks < KSPLIT; ++ks)
        v += Spart[ks * stride + (size_t)rowg * CC + t];

    __shared__ float red[256];
    red[t] = v;
    __syncthreads();
    for (int s = 128; s > 0; s >>= 1) {
        if (t < s) red[t] = fmaxf(red[t], red[t + s]);
        __syncthreads();
    }
    const float mx = red[0];
    __syncthreads();
    const float e = __expf(v - mx);
    red[t] = e;
    __syncthreads();
    for (int s = 128; s > 0; s >>= 1) {
        if (t < s) red[t] += red[t + s];
        __syncthreads();
    }
    const float m = e / red[0];
    // transposed store: MT[b][d=t][c]
    MT[((size_t)b << 16) + (size_t)t * CC + c] = (__bf16)m;
}

// ---------------------------------------------------------------------------
// Kernel 4: Out = gamma * (X @ M) + X   (K = 256)
// one wave per (16-row n-tile, 128-channel d-group): A reused over 8 accs
// total waves = 8 * 1024 * 2 = 16384 -> 2048 blocks x 256 thr
// ---------------------------------------------------------------------------
__global__ void ca_out_wmma(const float* __restrict__ x,
                            const __bf16* __restrict__ MT,
                            const float* __restrict__ gamma_p,
                            float* __restrict__ out) {
    const int gwave = (blockIdx.x * blockDim.x + threadIdx.x) >> 5; // 0..16383
    const int lane  = threadIdx.x & 31;
    const int b     = gwave >> 11;
    const int rem   = gwave & 2047;
    const int ntile = rem >> 1;       // 0..1023
    const int dgrp  = rem & 1;        // 0..1 (128 channels each)
    const int n0    = ntile * 16;
    const int lo16  = lane & 15;
    const int hi    = lane >> 4;

    const float*  xb = x  + (size_t)b * NN * CC;
    const __bf16* Mb = MT + ((size_t)b << 16);
    const float*  pAbase = xb + (size_t)(n0 + lo16) * CC + hi * 16;

    v8f acc[8] = {};
    for (int k0 = 0; k0 < CC; k0 += 32) {
        const float* pA = pAbase + k0;
        float4 f0 = *(const float4*)(pA);
        float4 f1 = *(const float4*)(pA + 4);
        float4 f2 = *(const float4*)(pA + 8);
        float4 f3 = *(const float4*)(pA + 12);
        v16bf a;
        a[0]=(__bf16)f0.x; a[1]=(__bf16)f0.y; a[2]=(__bf16)f0.z; a[3]=(__bf16)f0.w;
        a[4]=(__bf16)f1.x; a[5]=(__bf16)f1.y; a[6]=(__bf16)f1.z; a[7]=(__bf16)f1.w;
        a[8]=(__bf16)f2.x; a[9]=(__bf16)f2.y; a[10]=(__bf16)f2.z; a[11]=(__bf16)f2.w;
        a[12]=(__bf16)f3.x; a[13]=(__bf16)f3.y; a[14]=(__bf16)f3.z; a[15]=(__bf16)f3.w;
        #pragma unroll
        for (int dg = 0; dg < 8; ++dg) {
            const int drow = dgrp * 128 + dg * 16 + lo16;   // output channel column
            v16bf bm = *(const v16bf*)(Mb + (size_t)drow * CC + k0 + hi * 16);
            acc[dg] = __builtin_amdgcn_wmma_f32_16x16x32_bf16(
                          false, a, false, bm, (short)0, acc[dg], false, false);
        }
    }

    const float g = *gamma_p;
    #pragma unroll
    for (int dg = 0; dg < 8; ++dg) {
        const int d0 = dgrp * 128 + dg * 16;
        #pragma unroll
        for (int r = 0; r < 8; ++r) {
            const int m = r + hi * 8;
            const size_t idx = ((size_t)b * NN + n0 + m) * CC + d0 + lo16;
            out[idx] = g * acc[dg][r] + x[idx];
        }
    }
}

// ---------------------------------------------------------------------------
// launcher
// ---------------------------------------------------------------------------
extern "C" void kernel_launch(void* const* d_in, const int* in_sizes, int n_in,
                              void* d_out, int out_size, void* d_ws, size_t ws_size,
                              hipStream_t stream) {
    const float* x     = (const float*)d_in[0];
    const float* gamma = (const float*)d_in[1];
    float* out = (float*)d_out;

    char* ws = (char*)d_ws;
    __bf16* XT    = (__bf16*)ws;                                  // 64 MiB
    float*  Spart = (float*)(ws + (size_t)BB * CC * NN * 2);      // 4 * 2 MiB
    __bf16* MT    = (__bf16*)(ws + (size_t)BB * CC * NN * 2
                                 + (size_t)KSPLIT * BB * CC * CC * 4); // 1 MiB

    dim3 gT(NN / 64, CC / 64, BB);
    ca_transpose_cvt<<<gT, 256, 0, stream>>>(x, XT);
    ca_gram_wmma<<<(BB * 8 * 4 * KSPLIT) / 8, 256, 0, stream>>>(XT, Spart);
    ca_softmax_rows<<<BB * CC, 256, 0, stream>>>(Spart, MT);
    ca_out_wmma<<<(BB * (NN / 16) * 2) / 8, 256, 0, stream>>>(x, MT, gamma, out);
}
